// TextCNN2_77180562309367
// MI455X (gfx1250) — matrix operand order repaired and checked by reading, not compiled, not run on hardware
//
#include <hip/hip_runtime.h>
#include <hip/hip_bf16.h>

typedef _Float16 v16h __attribute__((ext_vector_type(16)));
typedef _Float16 v8h  __attribute__((ext_vector_type(8)));
typedef float    v8f  __attribute__((ext_vector_type(8)));

#define BATCH 32768
#define LSEQ  8
#define EBDIN 128
#define NK    64
#define NC    10

// workspace layout (bytes)
#define WS_XQ_BYTES  ((size_t)BATCH * LSEQ * EBDIN * 2)       // 64 MB f16
#define W3_ELEMS     (NK * 3 * EBDIN)
#define W4_ELEMS     (NK * 4 * EBDIN)
#define W5_ELEMS     (NK * 5 * EBDIN)

// ---------------------------------------------------------------------------
// Kernel 0: convert conv weights f32 -> f16 (tiny, L2-resident afterwards)
// ---------------------------------------------------------------------------
__global__ void __launch_bounds__(256) k_cvtw(const float* __restrict__ src,
                                              _Float16* __restrict__ dst,
                                              int n) {
  int i = blockIdx.x * blockDim.x + threadIdx.x;
  if (i < n) dst[i] = (_Float16)src[i];
}

// ---------------------------------------------------------------------------
// Kernel 1: embeddings + fc1 + sign/argmax quantizer -> xq (f16) in workspace
// One thread per (b, l, chunk) ; chunk = 32 features reshaped [4,8].
// ---------------------------------------------------------------------------
__global__ void __launch_bounds__(256) k_quant(
    const int* __restrict__ x, const float* __restrict__ lenE,
    const float* __restrict__ ipdE, const float* __restrict__ fc1w,
    const float* __restrict__ fc1b, const float* __restrict__ S,
    const float* __restrict__ Hm, const float* __restrict__ Tm,
    const float* __restrict__ LUT, _Float16* __restrict__ xq) {
  int gid = blockIdx.x * blockDim.x + threadIdx.x;   // < B*L*4
  int chunk = gid & 3;
  int bl = gid >> 2;                                 // b*L + l
  int i0 = x[bl * 2 + 0];
  int i1 = x[bl * 2 + 1];

  float4 e4[8];
  const float4* l4 = (const float4*)(lenE + (size_t)i0 * 16);
  const float4* p4 = (const float4*)(ipdE + (size_t)i1 * 16);
#pragma unroll
  for (int q = 0; q < 4; ++q) e4[q] = l4[q];
#pragma unroll
  for (int q = 0; q < 4; ++q) e4[4 + q] = p4[q];

  _Float16* outp = xq + (size_t)bl * 128 + chunk * 32;

#pragma unroll
  for (int c = 0; c < 4; ++c) {
    // fc1: 8 outputs of this (chunk, c)
    float h[8];
#pragma unroll
    for (int d = 0; d < 8; ++d) {
      int f = chunk * 32 + c * 8 + d;
      const float4* w4 = (const float4*)(fc1w + (size_t)f * 32);
      float a = fc1b[f];
#pragma unroll
      for (int q = 0; q < 8; ++q) {
        float4 w = w4[q];
        a += e4[q].x * w.x + e4[q].y * w.y + e4[q].z * w.z + e4[q].w * w.w;
      }
      h[d] = a;
    }
    // z = h . S[c] - T[c] - 1e-4 ; s = sign(z)
    float s[15];
#pragma unroll
    for (int k = 0; k < 15; ++k) {
      float z = -Tm[c * 15 + k] - 1e-4f;
#pragma unroll
      for (int d = 0; d < 8; ++d) z += h[d] * S[(c * 8 + d) * 15 + k];
      s[k] = (z > 0.f) ? 1.f : ((z < 0.f) ? -1.f : 0.f);
    }
    // z2 = s . H ; argmax (first max wins)
    float best = -3.4e38f;
    int bi = 0;
#pragma unroll
    for (int j = 0; j < 16; ++j) {
      float a = 0.f;
#pragma unroll
      for (int k = 0; k < 15; ++k) a += s[k] * Hm[k * 16 + j];
      if (a > best) { best = a; bi = j; }
    }
    // hard LUT lookup -> 8 f16 values
    const float* lp = LUT + (c * 16 + bi) * 8;
    v8h qv;
#pragma unroll
    for (int d = 0; d < 8; ++d) qv[d] = (_Float16)lp[d];
    *(v8h*)(outp + c * 8) = qv;
  }
}

// ---------------------------------------------------------------------------
// Kernel 2: three convs as WMMA GEMMs.  2 waves/block, 16 samples per wave.
// A from async-staged LDS tile, B (pre-converted f16 weights) from global.
// ---------------------------------------------------------------------------
template <int K>
__device__ __forceinline__ void conv_block(const _Float16* __restrict__ sx,
                                           const _Float16* __restrict__ wh,
                                           const float* __restrict__ bias,
                                           float* __restrict__ feat, int tile,
                                           int lane, int ntile) {
  constexpr int P = LSEQ - K + 1;
  constexpr int CO = (K == 3) ? 0 : ((K == 4) ? 384 : 704);
  const int laneN = lane & 15;
  const int laneHi = lane >> 4;
  const int filt = ntile * 16 + laneN;

  float bv = bias[filt];
  v8f acc[P];
#pragma unroll
  for (int t = 0; t < P; ++t)
#pragma unroll
    for (int r = 0; r < 8; ++r) acc[t][r] = bv;

#pragma unroll
  for (int dt = 0; dt < K; ++dt) {
#pragma unroll
    for (int kt = 0; kt < 4; ++kt) {
      // B fragment: lane (N=filt) holds 16 consecutive e-values (K dim)
      const _Float16* wp =
          wh + (size_t)(filt * K + dt) * 128 + kt * 32 + laneHi * 16;
      v8h blo = *(const v8h*)wp;
      v8h bhi = *(const v8h*)(wp + 8);
      v16h bf = __builtin_shufflevector(blo, bhi, 0, 1, 2, 3, 4, 5, 6, 7, 8, 9,
                                        10, 11, 12, 13, 14, 15);
#pragma unroll
      for (int t = 0; t < P; ++t) {
        int l = t + dt;
        // A fragment, contiguous k order (same as B):
        // lanes 0-15: K = base+0..15 ; lanes 16-31: K = base+16..31
        const _Float16* ap = sx + (laneN * 1024 + l * 128 + kt * 32 + laneHi * 16);
        v8h alo = *(const v8h*)ap;
        v8h ahi = *(const v8h*)(ap + 8);
        v16h a = __builtin_shufflevector(alo, ahi, 0, 1, 2, 3, 4, 5, 6, 7, 8, 9,
                                         10, 11, 12, 13, 14, 15);
        acc[t] = __builtin_amdgcn_wmma_f32_16x16x32_f16(
            false, a, false, bf, (short)0, acc[t], false, false);
      }
    }
  }
  // relu + scatter into feat[b, CO + filt*P + t]
#pragma unroll
  for (int t = 0; t < P; ++t) {
#pragma unroll
    for (int r = 0; r < 8; ++r) {
      float v = acc[t][r];
      v = v > 0.f ? v : 0.f;
      int samp = tile + laneHi * 8 + r;
      feat[(size_t)samp * 960 + CO + filt * P + t] = v;
    }
  }
}

__global__ void __launch_bounds__(64) k_conv(
    const _Float16* __restrict__ xq, const _Float16* __restrict__ w3h,
    const float* __restrict__ b3, const _Float16* __restrict__ w4h,
    const float* __restrict__ b4, const _Float16* __restrict__ w5h,
    const float* __restrict__ b5, float* __restrict__ out) {
  __shared__ _Float16 sxq[2][16 * 1024];  // 64 KB: 2 waves x 16 samples x 1024
  const int wave = threadIdx.x >> 5;
  const int lane = threadIdx.x & 31;
  const int tile = blockIdx.x * 32 + wave * 16;

  // pull f16 weights toward this WGP (global_prefetch_b8)
  __builtin_prefetch(w3h, 0, 0);
  __builtin_prefetch(w4h, 0, 0);
  __builtin_prefetch(w5h, 0, 0);

  // stage 16 samples (32 KB) into this wave's LDS region via CDNA5
  // async memory->LDS ops (tracked by ASYNCcnt, no VGPR round-trip)
  {
    unsigned int ldsbase = (unsigned int)(uintptr_t)(&sxq[wave][0]);
    const char* gbase = (const char*)(xq + (size_t)tile * 1024);
    for (int i = lane; i < 2048; i += 32) {  // 2048 x 16B = 32 KB
      unsigned int la = ldsbase + (unsigned int)i * 16u;
      unsigned long long ga = (unsigned long long)(uintptr_t)(gbase + (size_t)i * 16);
      asm volatile("global_load_async_to_lds_b128 %0, %1, off"
                   :
                   : "v"(la), "v"(ga)
                   : "memory");
    }
    asm volatile("s_wait_asynccnt 0x0" ::: "memory");
  }
  __syncthreads();

  float* feat = out + (size_t)BATCH * NC;  // feat region after logits
  for (int ntile = 0; ntile < 4; ++ntile) {
    conv_block<3>(sxq[wave], w3h, b3, feat, tile, lane, ntile);
    conv_block<4>(sxq[wave], w4h, b4, feat, tile, lane, ntile);
    conv_block<5>(sxq[wave], w5h, b5, feat, tile, lane, ntile);
  }
}

// ---------------------------------------------------------------------------
// Kernel 3: fc2 (960 -> 10) + log_softmax.  One wave per sample.
// ---------------------------------------------------------------------------
__global__ void __launch_bounds__(256) k_fc2(
    const float* __restrict__ fc2w, const float* __restrict__ fc2b,
    float* __restrict__ out) {
  __shared__ float sw[NC * 960];
  __shared__ float sb[NC];
  for (int i = threadIdx.x; i < NC * 960; i += blockDim.x) sw[i] = fc2w[i];
  if (threadIdx.x < NC) sb[threadIdx.x] = fc2b[threadIdx.x];
  __syncthreads();

  const int wave = threadIdx.x >> 5;
  const int lane = threadIdx.x & 31;
  const int samp = blockIdx.x * 8 + wave;
  const float* f = out + (size_t)BATCH * NC + (size_t)samp * 960;

  float p[NC];
#pragma unroll
  for (int j = 0; j < NC; ++j) p[j] = 0.f;
  for (int e = lane; e < 960; e += 32) {
    float fv = f[e];
#pragma unroll
    for (int j = 0; j < NC; ++j) p[j] += fv * sw[j * 960 + e];
  }
#pragma unroll
  for (int off = 16; off > 0; off >>= 1) {
#pragma unroll
    for (int j = 0; j < NC; ++j) p[j] += __shfl_xor(p[j], off, 32);
  }
  if (lane == 0) {
    float lg[NC];
    float m = -3.4e38f;
#pragma unroll
    for (int j = 0; j < NC; ++j) {
      lg[j] = p[j] + sb[j];
      m = lg[j] > m ? lg[j] : m;
    }
    float se = 0.f;
#pragma unroll
    for (int j = 0; j < NC; ++j) se += expf(lg[j] - m);
    float lse = m + logf(se);
#pragma unroll
    for (int j = 0; j < NC; ++j) out[(size_t)samp * NC + j] = lg[j] - lse;
  }
}

// ---------------------------------------------------------------------------
extern "C" void kernel_launch(void* const* d_in, const int* in_sizes, int n_in,
                              void* d_out, int out_size, void* d_ws,
                              size_t ws_size, hipStream_t stream) {
  const int*   x     = (const int*)d_in[0];
  const float* lenE  = (const float*)d_in[1];
  const float* ipdE  = (const float*)d_in[2];
  const float* fc1w  = (const float*)d_in[3];
  const float* fc1b  = (const float*)d_in[4];
  const float* S     = (const float*)d_in[5];
  const float* Hm    = (const float*)d_in[6];
  const float* Tm    = (const float*)d_in[7];
  const float* LUT   = (const float*)d_in[8];
  const float* w3    = (const float*)d_in[9];
  const float* b3    = (const float*)d_in[10];
  const float* w4    = (const float*)d_in[11];
  const float* b4    = (const float*)d_in[12];
  const float* w5    = (const float*)d_in[13];
  const float* b5    = (const float*)d_in[14];
  const float* fc2w  = (const float*)d_in[15];
  const float* fc2b  = (const float*)d_in[16];
  float* out = (float*)d_out;

  // workspace layout
  _Float16* xq  = (_Float16*)d_ws;                       // 64 MB
  _Float16* w3h = (_Float16*)((char*)d_ws + WS_XQ_BYTES);
  _Float16* w4h = w3h + W3_ELEMS;
  _Float16* w5h = w4h + W4_ELEMS;

  // K0: convert conv weights to f16
  k_cvtw<<<(W3_ELEMS + 255) / 256, 256, 0, stream>>>(w3, w3h, W3_ELEMS);
  k_cvtw<<<(W4_ELEMS + 255) / 256, 256, 0, stream>>>(w4, w4h, W4_ELEMS);
  k_cvtw<<<(W5_ELEMS + 255) / 256, 256, 0, stream>>>(w5, w5h, W5_ELEMS);

  // K1: B*L*4 = 1,048,576 threads
  k_quant<<<4096, 256, 0, stream>>>(x, lenE, ipdE, fc1w, fc1b, S, Hm, Tm, LUT,
                                    xq);
  // K2: 32 samples per block (2 waves x 16)
  k_conv<<<BATCH / 32, 64, 0, stream>>>(xq, w3h, b3, w4h, b4, w5h, b5, out);
  // K3: 8 samples per block (8 waves)
  k_fc2<<<BATCH / 8, 256, 0, stream>>>(fc2w, fc2b, out);
}
